// ES_MOE_36197984371395
// MI455X (gfx1250) — matrix operand order repaired and checked by reading, not compiled, not run on hardware
//
#include <hip/hip_runtime.h>

typedef _Float16 h16;
typedef __attribute__((ext_vector_type(2)))  _Float16 h2;
typedef __attribute__((ext_vector_type(16))) _Float16 v16h;
typedef __attribute__((ext_vector_type(8)))  float    v8f;
typedef __attribute__((ext_vector_type(4)))  unsigned int u32x4;
typedef __attribute__((ext_vector_type(8)))  int      i32x8;
typedef __attribute__((ext_vector_type(4)))  int      i32x4;

#define C_    128
#define NP    6400              // H*W per image
#define TPI   400               // 16-pixel tiles per image
#define ACT_BYTES (6400UL * 12288UL)   // 6400 tiles * 3 experts * 4 kblocks * 1024B

#if defined(__gfx1250__) && __has_builtin(__builtin_amdgcn_tensor_load_to_lds) && \
    __has_builtin(__builtin_amdgcn_s_wait_tensorcnt)
#define HAVE_TDM 1
#else
#define HAVE_TDM 0
#endif

// ---------------------------------------------------------------------------
// Kernel R: routing network.  Per 16-pixel tile: h = relu(W1 @ x) via WMMA
// (M=16=Cr, K=128, N=16 pixels), logits = W2 @ h (VALU + half-wave shuffle),
// softmax over 3 experts -> wsr[t][e][n].
// ---------------------------------------------------------------------------
__global__ __launch_bounds__(256) void moe_routing(
    const float* __restrict__ x,  const float* __restrict__ w1,
    const float* __restrict__ b1, const float* __restrict__ w2,
    const float* __restrict__ b2, float* __restrict__ wsr)
{
  const int wave = threadIdx.x >> 5, lane = threadIdx.x & 31;
  const int t   = blockIdx.x * 8 + wave;          // 0..6399
  const int img = t / TPI, rem = t % TPI;
  const int row = rem / 5, cb = rem % 5;
  const int n = lane & 15, hi = lane >> 4;
  const int pix = row * 80 + cb * 16 + n;
  const float* xb = x + (size_t)img * C_ * NP;

  // A fragments of W1 (16x128, f32 -> f16).  Lane: M = n, K = kb*32 + 16*hi + element (contiguous order).
  v16h a[4];
#pragma unroll
  for (int kb = 0; kb < 4; ++kb) {
#pragma unroll
    for (int v = 0; v < 8; ++v) {
      const int kl = (hi ? 16 : 0) + 2 * v;
      const int K  = kb * 32 + kl;
      a[kb][2 * v]     = (h16)w1[n * C_ + K];
      a[kb][2 * v + 1] = (h16)w1[n * C_ + K + 1];
    }
  }

  // B fragments of x (K = cin, N = pixel), accumulate 4 WMMAs (K=128).
  v8f acc = {};
#pragma unroll
  for (int kb = 0; kb < 4; ++kb) {
    v16h bf;
#pragma unroll
    for (int v = 0; v < 8; ++v) {
      const int K = kb * 32 + (hi ? 16 : 0) + 2 * v;
      bf[2 * v]     = (h16)xb[(size_t)K * NP + pix];
      bf[2 * v + 1] = (h16)xb[(size_t)(K + 1) * NP + pix];
    }
    acc = __builtin_amdgcn_wmma_f32_16x16x32_f16(false, a[kb], false, bf,
                                                 (short)0, acc, false, false);
  }

  // bias + relu (rows: VGPR r -> M = r + 8*hi)
  float hcr[8];
#pragma unroll
  for (int r = 0; r < 8; ++r) {
    float hv = acc[r] + b1[r + 8 * hi];
    hcr[r] = hv > 0.f ? hv : 0.f;
  }

  // logits: 3x16 @ h (column n split across half-waves -> shfl_xor 16)
  float lg[3];
#pragma unroll
  for (int e = 0; e < 3; ++e) {
    float s = 0.f;
#pragma unroll
    for (int r = 0; r < 8; ++r) s += hcr[r] * w2[e * 16 + r + 8 * hi];
    s += __shfl_xor(s, 16, 32);
    lg[e] = s + b2[e];
  }
  const float mx = fmaxf(lg[0], fmaxf(lg[1], lg[2]));
  const float e0 = __expf(lg[0] - mx), e1 = __expf(lg[1] - mx),
              e2 = __expf(lg[2] - mx);
  const float inv = 1.f / (e0 + e1 + e2);
  if (!hi) {
    float* o = wsr + (size_t)t * 48;
    o[0 * 16 + n] = e0 * inv;
    o[1 * 16 + n] = e1 * inv;
    o[2 * 16 + n] = e2 * inv;
  }
}

// ---------------------------------------------------------------------------
// Kernel D: depthwise conv (k=3/5/7) + routing pre-scale + pack into WMMA
// B-fragment layout (f16).  Block = 8x16 spatial tile x 64 channels (half).
// ---------------------------------------------------------------------------
template <int KK>
__device__ __forceinline__ void dwacc(const _Float16* xs, const float* dwf,
                                      int tb, int lpy0, int lpx0, int cp0,
                                      float& f0, float& f1)
{
  f0 = 0.f; f1 = 0.f;
#pragma unroll 1
  for (int dy = 0; dy < KK; ++dy) {
#pragma unroll
    for (int dx = 0; dx < KK; ++dx) {
      const int py = lpy0 + dy - KK / 2, px = lpx0 + dx - KK / 2;
      const h2 xv = *(const h2*)(xs + ((py * 22 + px) << 6) + cp0);
      const float* wp = dwf + ((tb + dy * KK + dx) << 6) + cp0;
      f0 += (float)xv.x * wp[0];
      f1 += (float)xv.y * wp[1];
    }
  }
}

__global__ __launch_bounds__(256) void moe_depthwise(
    const float* __restrict__ x,
    const float* __restrict__ dw0, const float* __restrict__ dw1,
    const float* __restrict__ dw2,
    const float* __restrict__ wsr, char* __restrict__ wsact)
{
  __shared__ _Float16 xs[308 * 64];   // (8+6)x(16+6) halo tile, ch-contiguous
  __shared__ float    dwf[83 * 64];   // packed dw weights for this half

  const int tid  = threadIdx.x;
  const int s    = blockIdx.x >> 1;        // spatial block 0..799
  const int half = blockIdx.x & 1;         // channel half
  const int img  = s / 50, loc = s % 50;
  const int th = loc / 5, tw = loc % 5;
  const int row0 = th * 8, col0 = tw * 16;
  const int cin0 = half * 64;

  // stage x tile (+halo) as f16, zero-padded at image borders
  for (int i = tid; i < 308 * 64; i += 256) {
    const int c = i / 308, sp = i - c * 308;
    const int py = sp / 22, px = sp - py * 22;
    const int gr = row0 + py - 3, gc = col0 + px - 3;
    float v = 0.f;
    if (gr >= 0 && gr < 80 && gc >= 0 && gc < 80)
      v = x[(size_t)(img * C_ + cin0 + c) * NP + gr * 80 + gc];
    xs[(py * 22 + px) * 64 + c] = (_Float16)v;
  }
  // stage depthwise weights (tap-major, channel-contiguous)
  for (int j = tid; j < 83 * 64; j += 256) {
    const int tg = j >> 6, c = j & 63;
    float v;
    if (tg < 9)       v = dw0[(cin0 + c) * 9  + tg];
    else if (tg < 34) v = dw1[(cin0 + c) * 25 + (tg - 9)];
    else              v = dw2[(cin0 + c) * 49 + (tg - 34)];
    dwf[j] = v;
  }
  __syncthreads();

  const int wave = tid >> 5, lane = tid & 31;
  const int n = lane & 15, hi = lane >> 4;
  const int row = row0 + wave;
  const int t = img * TPI + row * 5 + tw;

  float re3[3];
#pragma unroll
  for (int e = 0; e < 3; ++e) re3[e] = wsr[(size_t)t * 48 + e * 16 + n];

  const int lpy0 = wave + 3, lpx0 = n + 3;

#pragma unroll
  for (int kb = 0; kb < 2; ++kb) {
    const int kbG = half * 2 + kb;
#pragma unroll
    for (int e = 0; e < 3; ++e) {
      unsigned int u[8];
#pragma unroll
      for (int v = 0; v < 8; ++v) {
        const int cp0 = kb * 32 + (hi ? 16 : 0) + 2 * v;
        float f0, f1;
        if (e == 0)      dwacc<3>(xs, dwf, 0,  lpy0, lpx0, cp0, f0, f1);
        else if (e == 1) dwacc<5>(xs, dwf, 9,  lpy0, lpx0, cp0, f0, f1);
        else             dwacc<7>(xs, dwf, 34, lpy0, lpx0, cp0, f0, f1);
        h2 hv;
        hv.x = (h16)(f0 * re3[e]);
        hv.y = (h16)(f1 * re3[e]);
        u[v] = __builtin_bit_cast(unsigned int, hv);
      }
      char* dst = wsact + ((((size_t)t * 3 + e) * 4 + kbG) << 10) + lane * 32;
      *(uint4*)(dst)      = make_uint4(u[0], u[1], u[2], u[3]);
      *(uint4*)(dst + 16) = make_uint4(u[4], u[5], u[6], u[7]);
    }
  }
}

// ---------------------------------------------------------------------------
// TDM: one descriptor copies a whole 12288B B-fragment tile global->LDS.
// D# group0: {count=1, lds_addr, global_addr, type=2}
// D# group1: {data_size=4B, tensor_dim0=3072, tensor_dim1=1, tile_dim0=3072,
//             tensor_dim0_stride=3072}
// ---------------------------------------------------------------------------
#if HAVE_TDM
__device__ __forceinline__ void tdm_load_tile(const char* gsrc, unsigned ldsOff)
{
  const unsigned long long ga = (unsigned long long)gsrc;
  u32x4 g0;
  g0[0] = 1u;                                              // count=1, user mode
  g0[1] = ldsOff;                                          // lds_addr (bytes)
  g0[2] = (unsigned)ga;                                    // global_addr[31:0]
  g0[3] = (unsigned)((ga >> 32) & 0x1FFFFFFu) | (2u << 30); // ga[56:32] | type=2
  i32x8 g1;
  g1[0] = 0x00020000;              // workgroup_mask=0, data_size=2 (4 bytes)
  g1[1] = (int)(3072u << 16);      // tensor_dim0[15:0] = 3072 dwords
  g1[2] = (int)(1u << 16);         // tensor_dim0[31:16]=0, tensor_dim1=1
  g1[3] = (int)(3072u << 16);      // tensor_dim1[31:16]=0, tile_dim0=3072
  g1[4] = 0;                       // tile_dim1=0 (unused), tile_dim2=0
  g1[5] = 3072;                    // tensor_dim0_stride[31:0]
  g1[6] = 0;
  g1[7] = 0;
  const i32x4 z4 = {0, 0, 0, 0};
#if __clang_major__ >= 23
  const i32x8 z8 = {0, 0, 0, 0, 0, 0, 0, 0};
  __builtin_amdgcn_tensor_load_to_lds(g0, g1, z4, z4, z8, 0);
#else
  __builtin_amdgcn_tensor_load_to_lds(g0, g1, z4, z4, 0);
#endif
}
#endif

// ---------------------------------------------------------------------------
// Kernel G: pointwise GEMM (12 WMMAs per tile: 3 experts x K=128) + BN + SiLU.
// Block = full cout (8 waves x 16) over 8 pixel tiles.  B fragments staged in
// LDS, double-buffered via the Tensor Data Mover when available.
// ---------------------------------------------------------------------------
__global__ __launch_bounds__(256) void moe_gemm(
    const char* __restrict__ wsact,
    const float* __restrict__ pw0, const float* __restrict__ pw1,
    const float* __restrict__ pw2,
    const float* __restrict__ gamma, const float* __restrict__ beta,
    const float* __restrict__ mean,  const float* __restrict__ var,
    float* __restrict__ out)
{
#if HAVE_TDM
  __shared__ float4 bst[2][768];       // 2 x 12KB: double-buffered B tiles
#else
  __shared__ float4 bst[1][768];
#endif
  __shared__ float bnsc[C_], bnsh[C_];

  const int tid  = threadIdx.x;
  const int wave = tid >> 5, lane = tid & 31;
  const int n = lane & 15, hi = lane >> 4;
  const int coutBase = wave * 16;

  if (tid < C_) {
    const float sc = gamma[tid] * rsqrtf(var[tid] + 1e-5f);
    bnsc[tid] = sc;
    bnsh[tid] = beta[tid] - mean[tid] * sc;
  }

  // A fragments of all 3 pointwise weight matrices, kept in VGPRs
  const float* pws[3] = {pw0, pw1, pw2};
  v16h af[3][4];
#pragma unroll
  for (int e = 0; e < 3; ++e) {
#pragma unroll
    for (int kb = 0; kb < 4; ++kb) {
#pragma unroll
      for (int v = 0; v < 8; ++v) {
        const int kl = (hi ? 16 : 0) + 2 * v;
        const int K  = kb * 32 + kl;
        const float* p = pws[e] + (coutBase + n) * C_ + K;
        af[e][kb][2 * v]     = (h16)p[0];
        af[e][kb][2 * v + 1] = (h16)p[1];
      }
    }
  }

#if HAVE_TDM
  const unsigned ldsB[2] = {(unsigned)(size_t)&bst[0][0],
                            (unsigned)(size_t)&bst[1][0]};
  if (wave == 0)   // TDM ignores EXEC; issue once (wave-uniform descriptor)
    tdm_load_tile(wsact + (size_t)(blockIdx.x * 8) * 12288, ldsB[0]);
#endif

  for (int i = 0; i < 8; ++i) {
    const int t = blockIdx.x * 8 + i;

#if HAVE_TDM
    if (wave == 0) {
      if (i < 7) {
        tdm_load_tile(wsact + (size_t)(t + 1) * 12288, ldsB[(i + 1) & 1]);
        __builtin_amdgcn_s_wait_tensorcnt(1);   // tile i done, i+1 in flight
      } else {
        __builtin_amdgcn_s_wait_tensorcnt(0);
      }
    }
    __syncthreads();                  // publish TDM-written LDS to all waves
    const float4* bb = bst[i & 1];
#else
    __syncthreads();                  // previous tile fully consumed
    const float4* src = (const float4*)(wsact + (size_t)t * 12288);
#pragma unroll
    for (int j = 0; j < 3; ++j) bst[0][j * 256 + tid] = src[j * 256 + tid];
    if (i < 7)
      __builtin_prefetch(wsact + (size_t)(t + 1) * 12288 + tid * 48, 0, 1);
    __syncthreads();
    const float4* bb = bst[0];
#endif

    // 3 independent accumulator chains (one per expert) -> WMMAs interleave
    v8f acc0 = {}, acc1 = {}, acc2 = {};
#pragma unroll
    for (int kb = 0; kb < 4; ++kb) {
      union { float4 f4[2]; v16h h; } b0, b1, b2;
      b0.f4[0] = bb[(0 * 4 + kb) * 64 + lane * 2];
      b0.f4[1] = bb[(0 * 4 + kb) * 64 + lane * 2 + 1];
      b1.f4[0] = bb[(1 * 4 + kb) * 64 + lane * 2];
      b1.f4[1] = bb[(1 * 4 + kb) * 64 + lane * 2 + 1];
      b2.f4[0] = bb[(2 * 4 + kb) * 64 + lane * 2];
      b2.f4[1] = bb[(2 * 4 + kb) * 64 + lane * 2 + 1];
      acc0 = __builtin_amdgcn_wmma_f32_16x16x32_f16(false, af[0][kb], false,
                                                    b0.h, (short)0, acc0,
                                                    false, false);
      acc1 = __builtin_amdgcn_wmma_f32_16x16x32_f16(false, af[1][kb], false,
                                                    b1.h, (short)0, acc1,
                                                    false, false);
      acc2 = __builtin_amdgcn_wmma_f32_16x16x32_f16(false, af[2][kb], false,
                                                    b2.h, (short)0, acc2,
                                                    false, false);
    }
    const v8f acc = acc0 + acc1 + acc2;

    // epilogue: BN (eval) + SiLU, fp32 store
    const int img = t / TPI, rem = t % TPI;
    const int pixI = (rem / 5) * 80 + (rem % 5) * 16 + n;
#pragma unroll
    for (int r = 0; r < 8; ++r) {
      const int c = coutBase + r + 8 * hi;
      float o = acc[r] * bnsc[c] + bnsh[c];
      o = o / (1.f + __expf(-o));
      out[(size_t)(img * C_ + c) * NP + pixI] = o;
    }
    __syncthreads();                  // all waves done with bb before reuse
  }
}

// ---------------------------------------------------------------------------
extern "C" void kernel_launch(void* const* d_in, const int* in_sizes, int n_in,
                              void* d_out, int out_size, void* d_ws,
                              size_t ws_size, hipStream_t stream)
{
  (void)in_sizes; (void)n_in; (void)out_size; (void)ws_size;
  // setup_inputs() dict order:
  const float* x     = (const float*)d_in[0];
  const float* r_w1  = (const float*)d_in[1];
  const float* r_b1  = (const float*)d_in[2];
  const float* r_w2  = (const float*)d_in[3];
  const float* r_b2  = (const float*)d_in[4];
  const float* gamma = (const float*)d_in[5];
  const float* beta  = (const float*)d_in[6];
  const float* mean  = (const float*)d_in[7];
  const float* var   = (const float*)d_in[8];
  const float* dw0   = (const float*)d_in[9];
  const float* pw0   = (const float*)d_in[10];
  const float* dw1   = (const float*)d_in[11];
  const float* pw1   = (const float*)d_in[12];
  const float* dw2   = (const float*)d_in[13];
  const float* pw2   = (const float*)d_in[14];
  float* out = (float*)d_out;

  char*  wsact = (char*)d_ws;
  float* wsr   = (float*)(wsact + ACT_BYTES);

  moe_routing<<<800, 256, 0, stream>>>(x, r_w1, r_b1, r_w2, r_b2, wsr);
  moe_depthwise<<<1600, 256, 0, stream>>>(x, dw0, dw1, dw2, wsr, wsact);
  moe_gemm<<<800, 256, 0, stream>>>(wsact, pw0, pw1, pw2,
                                    gamma, beta, mean, var, out);
}
